// AttentionSeq2Seq_65704409694660
// MI455X (gfx1250) — compile-verified, matrix operand order checked by reading, not run on hardware
//
#include <hip/hip_runtime.h>
#include <hip/hip_bf16.h>

// ---------------------------------------------------------------------------
// AttentionSeq2Seq (B=64, S=T=512, E=512, H=1024, O=15) for MI455X / gfx1250.
//
//  * degenerate softmax (size-1 axis) => context[b] = sum_s enc_h[b,s],
//    accumulated in registers during the encoder scan.
//  * tiny vocabularies => input GEMMs become token-indexed gate tables.
//  * recurrence: 1024 sequential steps of [64,1024]x[1024,4096] f16 WMMA
//    (v_wmma_f32_16x16x32_f16, f32 accumulate).
//  * 32 persistent WGs; each WG's 256 KB W_hh slice is TDM-loaded into LDS
//    once (padded rows, bank-conflict-free-ish) and served via ds_load_b128
//    for all 512 steps; h is exchanged via a double-buffered global f16
//    buffer with one agent-scope atomic grid barrier per step.
// ---------------------------------------------------------------------------

namespace {

constexpr int B_   = 64;
constexpr int SEQ_ = 512;
constexpr int E_   = 512;
constexpr int H_   = 1024;
constexpr int G4_  = 4096;   // 4*H
constexpr int O_   = 15;
constexpr int NWG_ = 32;     // persistent workgroups in the scan kernels

// LDS layout for the W slice: 128 rows (4 gates x 32 h-cols) of 1024 f16,
// padded +16B per 512 f16 => row stride 2080B (stride % 64 banks = 8 dwords,
// 16B-aligned b128 chunks; matches TDM pad_interval=256dw / pad_amount=4dw).
constexpr int SW_ROW_BYTES = 2080;
constexpr int SW_BYTES     = 128 * SW_ROW_BYTES;     // 266240
constexpr int SMEM_BYTES   = SW_BYTES + B_ * (int)sizeof(int);

typedef float        v8f   __attribute__((ext_vector_type(8)));
typedef _Float16     v16h  __attribute__((ext_vector_type(16)));
typedef unsigned int u32x4 __attribute__((ext_vector_type(4)));
typedef int          i32x4 __attribute__((ext_vector_type(4)));
typedef int          i32x8 __attribute__((ext_vector_type(8)));

union HFrag { v16h v; uint4 u[2]; };

__device__ __forceinline__ float sigf(float x)      { return 1.0f / (1.0f + __expf(-x)); }
__device__ __forceinline__ float tanh_fast(float x) { return 2.0f / (1.0f + __expf(-2.0f * x)) - 1.0f; }

// byte offset of f16 element k inside a padded LDS W row
__device__ __forceinline__ int swOff(int k) { return 2 * k + ((k & 512) ? 16 : 0); }

// ---------------- prep kernels --------------------------------------------

__global__ void zero_init_kernel(_Float16* hBuf, unsigned* counters, int nH) {
  int i = blockIdx.x * blockDim.x + threadIdx.x;
  if (i < nH) hBuf[i] = (_Float16)0.0f;
  if (i < 2)  counters[i] = 0u;
}

__global__ void convert_f16_kernel(const float* __restrict__ src,
                                   _Float16* __restrict__ dst, int n) {
  int i = blockIdx.x * blockDim.x + threadIdx.x;
  if (i < n) dst[i] = (_Float16)src[i];
}

// tab[v][g] = dot(emb[v][0:E], Wih[g][0:E]) + bih[g] + bhh[g]
__global__ void build_tab_kernel(const float* __restrict__ emb,
                                 const float* __restrict__ Wih, int wihStride,
                                 const float* __restrict__ bih,
                                 const float* __restrict__ bhh,
                                 float* __restrict__ tab, int V) {
  int idx = blockIdx.x * blockDim.x + threadIdx.x;
  if (idx >= V * G4_) return;
  int v = idx >> 12;
  int g = idx & (G4_ - 1);
  const float* er = emb + (size_t)v * E_;
  const float* wr = Wih + (size_t)g * wihStride;
  float s = 0.f;
  #pragma unroll 4
  for (int k = 0; k < E_; k += 4) {
    float4 a = *(const float4*)(er + k);
    float4 w = *(const float4*)(wr + k);
    s += a.x * w.x + a.y * w.y + a.z * w.z + a.w * w.w;
  }
  tab[idx] = s + bih[g] + bhh[g];
}

// ctxG[b][g] = dot(ctx[b][:], dec_W_ih[g][E : E+H])   (constant attention ctx)
__global__ void ctx_gates_kernel(const float* __restrict__ ctx,
                                 const float* __restrict__ decWih,
                                 float* __restrict__ ctxG) {
  int idx = blockIdx.x * blockDim.x + threadIdx.x;
  if (idx >= B_ * G4_) return;
  int b = idx >> 12;
  int g = idx & (G4_ - 1);
  const float* cr = ctx + (size_t)b * H_;
  const float* wr = decWih + (size_t)g * (E_ + H_) + E_;
  float s = 0.f;
  #pragma unroll 4
  for (int k = 0; k < H_; k += 4) {
    float4 a = *(const float4*)(cr + k);
    float4 w = *(const float4*)(wr + k);
    s += a.x * w.x + a.y * w.y + a.z * w.z + a.w * w.w;
  }
  ctxG[idx] = s;
}

// ---------------- grid barrier (agent scope, monotonic counter) ------------

__device__ __forceinline__ void grid_barrier(unsigned* bar, unsigned target) {
  __syncthreads();
  if (threadIdx.x == 0) {
    __hip_atomic_fetch_add(bar, 1u, __ATOMIC_ACQ_REL, __HIP_MEMORY_SCOPE_AGENT);
    while (__hip_atomic_load(bar, __ATOMIC_ACQUIRE, __HIP_MEMORY_SCOPE_AGENT) < target)
      __builtin_amdgcn_s_sleep(1);
  }
  __syncthreads();
}

// ---------------- persistent LSTM scan (encoder: IS_DEC=0, decoder: 1) -----

template <int IS_DEC>
__global__ __launch_bounds__(256, 1)
void rnn_scan_kernel(const int* __restrict__ tokens,        // [64,512]
                     const float* __restrict__ gateTab,     // [V,4096]
                     const float* __restrict__ ctxG,        // [64,4096] (dec)
                     const _Float16* __restrict__ Wb,       // [4096,1024] f16
                     _Float16* hBuf,                        // [2,64,1024] f16
                     float* __restrict__ cBuf,              // [64,1024]
                     float* __restrict__ ctxBuf,            // [64,1024] (enc)
                     const float* __restrict__ outW,        // [15,1024]
                     const float* __restrict__ outB,        // [15]
                     float* __restrict__ out,               // [64,512,15]
                     unsigned* bar) {
  extern __shared__ char smem[];
  _Float16* sW   = (_Float16*)smem;               // padded W slice, 128 rows
  int*      sTok = (int*)(smem + SW_BYTES);

  const int tid  = threadIdx.x;
  const int wg   = blockIdx.x;                   // 0..31 : h-cols [32wg,32wg+32)
  const int lane = tid & 31;
  const int wave = tid >> 5;                     // 0..7
  const int m0   = (wave & 3) * 16;              // batch-row tile base
  const int hct  = wave >> 2;                    // hc tile within WG (0..1)
  const int hcb  = wg * 32 + hct * 16;           // h-column tile base
  const int lh   = lane >> 4;                    // K-half select (ISA 7.12.2)
  const int ln   = lane & 15;                    // M (A) / N (B,C/D) coord

  // -------- one-time preload of this WG's W_hh slice into LDS -------------
  // 4 tiles (one per gate): W rows [g*1024 + 32wg, +32), padded into LDS.
#if __has_builtin(__builtin_amdgcn_tensor_load_to_lds)
  if (wave == 0) {                // one wave issues 4 TDM descriptors
    #pragma unroll
    for (int g = 0; g < 4; ++g) {
      unsigned long long ga =
          (unsigned long long)(const void*)(Wb + (size_t)(g * H_ + wg * 32) * H_);
      unsigned la = (unsigned)(size_t)(void*)sW + (unsigned)(g * 32 * SW_ROW_BYTES);
      u32x4 g0 = { 1u,                                   // count=1 (valid D#)
                   la,                                   // lds_addr [63:32]
                   (unsigned)ga,                         // global_addr [95:64]
                   ((unsigned)(ga >> 32) & 0x01FFFFFFu)  // global_addr [120:96]
                     | (2u << 30) };                     // type=2 ("image")
      i32x8 g1;
      g1[0] = (1 << 16)      // data_size = 2 bytes
            | (1 << 20)      // pad_enable (LDS row padding)
            | (7 << 22)      // pad_interval: 256 dwords
            | (3 << 25);     // pad_amount:   4 dwords (16B)
      g1[1] = (int)(1024u << 16);  // tensor_dim0 = 1024 (low 16 @ [63:48])
      g1[2] = (int)(32u << 16);    // tensor_dim0 hi=0; tensor_dim1 = 32
      g1[3] = (int)(1024u << 16);  // tensor_dim1 hi=0; tile_dim0 = 1024
      g1[4] = 32;                  // tile_dim1 = 32; tile_dim2 = 0
      g1[5] = 1024;                // tensor_dim0_stride = 1024 (low 32)
      g1[6] = 0;
      g1[7] = 0;
      i32x4 z4 = { 0, 0, 0, 0 };
#if defined(__clang_major__) && (__clang_major__ <= 22)
      __builtin_amdgcn_tensor_load_to_lds(g0, g1, z4, z4, 0);
#else
      i32x8 z8 = { 0, 0, 0, 0, 0, 0, 0, 0 };
      __builtin_amdgcn_tensor_load_to_lds(g0, g1, z4, z4, z8, 0);
#endif
    }
    __builtin_amdgcn_s_wait_tensorcnt(0);
  }
#else
  for (int cc = tid; cc < 128 * 128; cc += 256) {   // 8-f16 chunks
    int lr = cc >> 7;                // local row 0..127
    int kc = (cc & 127) * 8;         // k chunk base
    int g  = lr >> 5, rr = lr & 31;
    uint4 v = *(const uint4*)(Wb + (size_t)(g * H_ + wg * 32 + rr) * H_ + kc);
    *(uint4*)((char*)sW + (size_t)lr * SW_ROW_BYTES + swOff(kc)) = v;
  }
#endif
  __syncthreads();

  // LDS base pointers for this wave's B rows (gate g, column hcb+ln)
  const char* bBase[4];
  #pragma unroll
  for (int g = 0; g < 4; ++g)
    bBase[g] = (const char*)sW + (size_t)(g * 32 + hct * 16 + ln) * SW_ROW_BYTES;

  float c[8], ctx[8];
  #pragma unroll
  for (int r = 0; r < 8; ++r) {
    const int m  = m0 + r + 8 * lh;              // C/D layout: VGPR r -> M
    const int hc = hcb + ln;
    c[r]   = IS_DEC ? cBuf[m * H_ + hc] : 0.f;
    ctx[r] = 0.f;
  }

  for (int s = 0; s < SEQ_; ++s) {
    if (tid < B_) sTok[tid] = tokens[tid * SEQ_ + s];
    __syncthreads();

    // Accumulators seeded with the precomputed input-gate contribution.
    v8f acc[4];
    #pragma unroll
    for (int g = 0; g < 4; ++g) {
      #pragma unroll
      for (int r = 0; r < 8; ++r) {
        const int m   = m0 + r + 8 * lh;
        const int col = g * H_ + hcb + ln;
        float v = gateTab[sTok[m] * G4_ + col];
        if (IS_DEC) v += ctxG[m * G4_ + col];
        acc[g][r] = v;
      }
    }

    const _Float16* hCur = hBuf + (size_t)(s & 1) * (B_ * H_);
    _Float16*       hNxt = hBuf + (size_t)((s + 1) & 1) * (B_ * H_);
    const _Float16* aRow = hCur + (size_t)(m0 + ln) * H_;     // A: lane -> M

    // gates[16, 4H-slice] += h[16,1024] @ W_hh^T : 32 k-steps x 4 WMMAs,
    // A from global (L2-resident h exchange), B from LDS.
    #pragma unroll 4
    for (int kk = 0; kk < H_ / 32; ++kk) {
      const int k0 = kk * 32 + lh * 8;   // K chunks {k0..k0+7, k0+16..k0+23}
      HFrag a;
      a.u[0] = *(const uint4*)(aRow + k0);
      a.u[1] = *(const uint4*)(aRow + k0 + 16);
      __builtin_prefetch(aRow + k0 + 128, 0, 3);
      const int o0 = swOff(k0);
      const int o1 = swOff(k0 + 16);
      #pragma unroll
      for (int g = 0; g < 4; ++g) {
        HFrag b;
        b.u[0] = *(const uint4*)(bBase[g] + o0);
        b.u[1] = *(const uint4*)(bBase[g] + o1);
        acc[g] = __builtin_amdgcn_wmma_f32_16x16x32_f16(
            false, a.v, false, b.v, (short)0, acc[g], false, false);
      }
    }

    // Fused LSTM nonlinearity (gate order i,f,g,o) + h store (f16).
    #pragma unroll
    for (int r = 0; r < 8; ++r) {
      const float iv = sigf(acc[0][r]);
      const float fv = sigf(acc[1][r]);
      const float gv = tanh_fast(acc[2][r]);
      const float ov = sigf(acc[3][r]);
      c[r] = fv * c[r] + iv * gv;
      const float h = ov * tanh_fast(c[r]);
      if (!IS_DEC) ctx[r] += h;   // degenerate attention: ctx = sum_s h
      const int m  = m0 + r + 8 * lh;
      const int hc = hcb + ln;
      hNxt[m * H_ + hc] = (_Float16)h;
    }

    grid_barrier(bar, (unsigned)NWG_ * (unsigned)(s + 1));

    if (IS_DEC) {
      // out[b][s][:] = h_new[b] @ out_W^T + out_b ; WG w handles b = 2w, 2w+1.
      for (int j = wave; j < 2 * O_; j += 8) {
        const int b = 2 * wg + (j >= O_ ? 1 : 0);
        const int o = (j >= O_) ? (j - O_) : j;
        const _Float16* hr = hNxt + (size_t)b * H_;
        const float*    wr = outW + (size_t)o * H_;
        float sum = 0.f;
        for (int k = lane; k < H_; k += 32)
          sum += (float)hr[k] * wr[k];
        #pragma unroll
        for (int off = 16; off > 0; off >>= 1)
          sum += __shfl_xor(sum, off, 32);
        if (lane == 0) out[((size_t)b * SEQ_ + s) * O_ + o] = sum + outB[o];
      }
    }
  }

  if (!IS_DEC) {
    #pragma unroll
    for (int r = 0; r < 8; ++r) {
      const int m  = m0 + r + 8 * lh;
      const int hc = hcb + ln;
      cBuf[m * H_ + hc]   = c[r];
      ctxBuf[m * H_ + hc] = ctx[r];
    }
  }
}

} // namespace

// ---------------------------------------------------------------------------

extern "C" void kernel_launch(void* const* d_in, const int* in_sizes, int n_in,
                              void* d_out, int out_size, void* d_ws, size_t ws_size,
                              hipStream_t stream) {
  (void)in_sizes; (void)n_in; (void)out_size; (void)ws_size;

  const int*   encTok = (const int*)  d_in[0];
  const int*   decTok = (const int*)  d_in[1];
  const float* encEmb = (const float*)d_in[2];
  const float* encWih = (const float*)d_in[3];
  const float* encWhh = (const float*)d_in[4];
  const float* encBih = (const float*)d_in[5];
  const float* encBhh = (const float*)d_in[6];
  const float* decEmb = (const float*)d_in[7];
  const float* decWih = (const float*)d_in[8];
  const float* decWhh = (const float*)d_in[9];
  const float* decBih = (const float*)d_in[10];
  const float* decBhh = (const float*)d_in[11];
  const float* outW   = (const float*)d_in[12];
  const float* outB   = (const float*)d_in[13];
  float* out = (float*)d_out;

  // Workspace carve (~20 MB total).
  char* p = (char*)d_ws;
  auto carve = [&p](size_t bytes) -> void* {
    void* r = (void*)p;
    p += (bytes + 255) & ~(size_t)255;
    return r;
  };
  _Float16* wEnc   = (_Float16*)carve((size_t)G4_ * H_ * 2);       // 8 MB
  _Float16* wDec   = (_Float16*)carve((size_t)G4_ * H_ * 2);       // 8 MB
  float*    encTab = (float*)   carve((size_t)64 * G4_ * 4);       // 1 MB
  float*    decTab = (float*)   carve((size_t)16 * G4_ * 4);       // 256 KB
  float*    ctxBuf = (float*)   carve((size_t)B_ * H_ * 4);        // 256 KB
  float*    ctxG   = (float*)   carve((size_t)B_ * G4_ * 4);       // 1 MB
  float*    cBuf   = (float*)   carve((size_t)B_ * H_ * 4);        // 256 KB
  _Float16* hBuf   = (_Float16*)carve((size_t)2 * B_ * H_ * 2);    // 256 KB
  unsigned* bars   = (unsigned*)carve(256);

  const int nW = G4_ * H_;
  const int nH = 2 * B_ * H_;

  zero_init_kernel<<<(nH + 255) / 256, 256, 0, stream>>>(hBuf, bars, nH);
  convert_f16_kernel<<<(nW + 255) / 256, 256, 0, stream>>>(encWhh, wEnc, nW);
  convert_f16_kernel<<<(nW + 255) / 256, 256, 0, stream>>>(decWhh, wDec, nW);
  build_tab_kernel<<<(64 * G4_ + 255) / 256, 256, 0, stream>>>(
      encEmb, encWih, E_, encBih, encBhh, encTab, 64);
  build_tab_kernel<<<(16 * G4_ + 255) / 256, 256, 0, stream>>>(
      decEmb, decWih, E_ + H_, decBih, decBhh, decTab, 16);

  rnn_scan_kernel<0><<<NWG_, 256, SMEM_BYTES, stream>>>(
      encTok, encTab, nullptr, wEnc, hBuf, cBuf, ctxBuf,
      nullptr, nullptr, nullptr, bars + 0);

  ctx_gates_kernel<<<(B_ * G4_ + 255) / 256, 256, 0, stream>>>(ctxBuf, decWih, ctxG);

  rnn_scan_kernel<1><<<NWG_, 256, SMEM_BYTES, stream>>>(
      decTok, decTab, ctxG, wDec, hBuf, cBuf, nullptr,
      outW, outB, out, bars + 1);
}
